// EnvSpecificHead_57028575756791
// MI455X (gfx1250) — compile-verified, matrix operand order checked by reading, not run on hardware
//
#include <hip/hip_runtime.h>
#include <hip/hip_bf16.h>

// ---------------------------------------------------------------------------
// Problem constants (from reference): B=8192, D=1024, H=2048, A=1024, E=4
// ---------------------------------------------------------------------------
#define BB 8192
#define DD 1024
#define HH 2048
#define AA 1024
#define EE 4

#define KPAD 40           // bf16 elements per LDS row (80B = 20 banks; rows 0..15 conflict-free)
#define TILE_M 128
#define TILE_N 128
#define TILE_K 32
#define MAX_MTILES (BB / TILE_M + EE)   // 68: worst-case ragged env segments

typedef __attribute__((ext_vector_type(16))) __bf16 v16bf;
typedef __attribute__((ext_vector_type(8)))  float  v8f;

union Frag16 {            // one 16x32 (A) or 32x16 (B) bf16 fragment per lane
    v16bf v;
    uint4 u4[2];
};

__device__ __forceinline__ __bf16 f2bf(float f) { return (__bf16)f; }

// ---------------------------------------------------------------------------
// Workspace layout (bytes):
//   [0    ..  16)  u32 counts[4]
//   [16   ..  36)  u32 offs[5]
//   [64   ..  80)  u32 cursors[4]
//   [256  .. 256+4*B)          int32 idx[B]   (sorted token indices)
//   [33024.. 33024+2*B*H)      bf16  hid[B][H] (sorted order)
// ---------------------------------------------------------------------------
#define WS_COUNTS 0
#define WS_OFFS   16
#define WS_CURS   64
#define WS_IDX    256
#define WS_HID    33024

__global__ void k_zero(unsigned* ws_u32) {
    if (threadIdx.x < EE) ws_u32[WS_COUNTS / 4 + threadIdx.x] = 0u;
}

__global__ void k_count(const long long* __restrict__ env, unsigned* ws_u32) {
    int i = blockIdx.x * blockDim.x + threadIdx.x;
    if (i < BB) {
        int e = (int)env[i];
        atomicAdd(&ws_u32[WS_COUNTS / 4 + e], 1u);
    }
}

__global__ void k_scan(unsigned* ws_u32) {
    if (threadIdx.x == 0) {
        unsigned run = 0;
        for (int e = 0; e < EE; ++e) {
            ws_u32[WS_OFFS / 4 + e] = run;
            ws_u32[WS_CURS / 4 + e] = run;
            run += ws_u32[WS_COUNTS / 4 + e];
        }
        ws_u32[WS_OFFS / 4 + EE] = run;   // == BB
    }
}

__global__ void k_scatter(const long long* __restrict__ env, unsigned* ws_u32,
                          int* __restrict__ idx) {
    int i = blockIdx.x * blockDim.x + threadIdx.x;
    if (i < BB) {
        int e = (int)env[i];
        unsigned pos = atomicAdd(&ws_u32[WS_CURS / 4 + e], 1u);
        idx[pos] = i;
    }
}

// ---------------------------------------------------------------------------
// Map blockIdx.x -> (env, m0, valid) over ragged per-env segments.
// Uniform across the block, so EXEC stays all-ones in the WMMA region.
// ---------------------------------------------------------------------------
__device__ __forceinline__ bool map_tile(const unsigned* offs, int t,
                                         int& e_out, int& m0_out, int& valid_out) {
    for (int e = 0; e < EE; ++e) {
        int s  = (int)offs[e];
        int en = (int)offs[e + 1];
        int tiles = (en - s + TILE_M - 1) >> 7;
        if (t < tiles) {
            int m0 = s + t * TILE_M;
            e_out = e; m0_out = m0;
            valid_out = min(TILE_M, en - m0);
            return true;
        }
        t -= tiles;
    }
    return false;
}

__device__ __forceinline__ v8f wmma_bf16(const Frag16& a, const Frag16& b, v8f c) {
    return __builtin_amdgcn_wmma_f32_16x16x32_bf16(
        /*neg_a=*/false, a.v, /*neg_b=*/false, b.v,
        /*c_mod=*/(short)0, c, /*reuse_a=*/false, /*reuse_b=*/false);
}

// ---------------------------------------------------------------------------
// GEMM1: hid[sorted] = relu( h[idx] @ W1[e] + b1[e] )
// Double-buffered LDS; global loads staged in registers one K-step ahead.
// ---------------------------------------------------------------------------
__global__ void __launch_bounds__(256)
k_gemm1(const float* __restrict__ h, const float* __restrict__ W1,
        const float* __restrict__ b1, const unsigned* __restrict__ offs,
        const int* __restrict__ idx, __bf16* __restrict__ hid) {
    __shared__ __bf16 As[2][TILE_M * KPAD];   // [m][k]
    __shared__ __bf16 Bs[2][TILE_N * KPAD];   // [n][k] (transposed)

    int e, m0, valid;
    if (!map_tile(offs, blockIdx.x, e, m0, valid)) return;
    const int n0 = blockIdx.y * TILE_N;
    const float* W = W1 + (size_t)e * DD * HH;

    const int lane = threadIdx.x & 31;
    const int wave = threadIdx.x >> 5;
    const int half = lane >> 4;
    const int r    = lane & 15;
    const int ksel = half * 8;

    // Precompute per-thread staging pointers (A row gather is kk-invariant).
    const float* arow[4];
    int arow_lds[4], asg[4];
    #pragma unroll
    for (int it = 0; it < 4; ++it) {
        int slot = threadIdx.x + it * 256;            // 1024 slots: 128 rows x 8 float4
        int row  = slot >> 3, seg = slot & 7;
        int srow = m0 + (row < valid ? row : valid - 1);
        arow[it]     = h + (size_t)idx[srow] * DD + seg * 4;
        arow_lds[it] = row * KPAD + seg * 4;
        asg[it]      = 0;
    }
    const float* brow[4];
    int bk[4], bn[4];
    #pragma unroll
    for (int it = 0; it < 4; ++it) {
        int slot = threadIdx.x + it * 256;            // 1024 slots: 32 k x 32 float4
        bk[it] = slot >> 5;
        bn[it] = (slot & 31) * 4;
        brow[it] = W + (size_t)bk[it] * HH + n0 + bn[it];
    }

    float4 fa[4], fb[4];
    auto load_stage = [&](int kk) {
        #pragma unroll
        for (int it = 0; it < 4; ++it) fa[it] = *(const float4*)(arow[it] + kk);
        #pragma unroll
        for (int it = 0; it < 4; ++it) fb[it] = *(const float4*)(brow[it] + (size_t)kk * HH);
    };
    auto store_stage = [&](int buf) {
        #pragma unroll
        for (int it = 0; it < 4; ++it) {
            __bf16* d = &As[buf][arow_lds[it]];
            d[0] = f2bf(fa[it].x); d[1] = f2bf(fa[it].y);
            d[2] = f2bf(fa[it].z); d[3] = f2bf(fa[it].w);
        }
        #pragma unroll
        for (int it = 0; it < 4; ++it) {
            Bs[buf][(bn[it] + 0) * KPAD + bk[it]] = f2bf(fb[it].x);
            Bs[buf][(bn[it] + 1) * KPAD + bk[it]] = f2bf(fb[it].y);
            Bs[buf][(bn[it] + 2) * KPAD + bk[it]] = f2bf(fb[it].z);
            Bs[buf][(bn[it] + 3) * KPAD + bk[it]] = f2bf(fb[it].w);
        }
    };

    v8f acc[8] = {};

    load_stage(0);
    store_stage(0);
    int cur = 0;
    for (int kk = 0; kk < DD; kk += TILE_K) {
        const bool more = (kk + TILE_K) < DD;
        if (more) load_stage(kk + TILE_K);            // overlap with compute below
        __syncthreads();                              // publish buf[cur]

        Frag16 a;
        a.u4[0] = *(const uint4*)(&As[cur][(wave * 16 + r) * KPAD + ksel]);
        a.u4[1] = *(const uint4*)(&As[cur][(wave * 16 + r) * KPAD + 16 + ksel]);
        #pragma unroll
        for (int ns = 0; ns < 8; ++ns) {
            Frag16 b;
            b.u4[0] = *(const uint4*)(&Bs[cur][(ns * 16 + r) * KPAD + ksel]);
            b.u4[1] = *(const uint4*)(&Bs[cur][(ns * 16 + r) * KPAD + 16 + ksel]);
            acc[ns] = wmma_bf16(a, b, acc[ns]);
        }
        if (more) store_stage(cur ^ 1);
        cur ^= 1;
    }

    // epilogue: bias + ReLU, store bf16 hid in sorted order
    #pragma unroll
    for (int ns = 0; ns < 8; ++ns) {
        int ncol = n0 + ns * 16 + r;
        float bias = b1[(size_t)e * HH + ncol];
        #pragma unroll
        for (int j = 0; j < 8; ++j) {
            int ml = wave * 16 + half * 8 + j;        // C/D layout: VGPR j -> row j(+8)
            if (ml < valid) {
                float v = acc[ns][j] + bias;
                v = v > 0.f ? v : 0.f;
                hid[(size_t)(m0 + ml) * HH + ncol] = f2bf(v);
            }
        }
    }
}

// ---------------------------------------------------------------------------
// GEMM2: out[idx] = hid[sorted] @ W2[e] + b2[e]
// A tile copied with GLOBAL_LOAD_ASYNC_TO_LDS_B128 (ASYNCcnt-tracked DMA),
// B tile register-staged with fp32->bf16 convert; double-buffered LDS.
// ---------------------------------------------------------------------------
__global__ void __launch_bounds__(256)
k_gemm2(const __bf16* __restrict__ hid, const float* __restrict__ W2,
        const float* __restrict__ b2, const unsigned* __restrict__ offs,
        const int* __restrict__ idx, float* __restrict__ out) {
    __shared__ __bf16 As[2][TILE_M * KPAD];
    __shared__ __bf16 Bs[2][TILE_N * KPAD];

    int e, m0, valid;
    if (!map_tile(offs, blockIdx.x, e, m0, valid)) return;
    const int n0 = blockIdx.y * TILE_N;
    const float* W = W2 + (size_t)e * HH * AA;

    const int lane = threadIdx.x & 31;
    const int wave = threadIdx.x >> 5;
    const int half = lane >> 4;
    const int r    = lane & 15;
    const int ksel = half * 8;

    // A staging: 512 slots = 128 rows x 4 x (16B = 8 bf16); per-lane async DMA.
    const __bf16* agsrc[2];
    int alds[2];
    #pragma unroll
    for (int it = 0; it < 2; ++it) {
        int slot = threadIdx.x + it * 256;
        int row = slot >> 2, seg = slot & 3;
        int srow = m0 + (row < valid ? row : valid - 1);
        agsrc[it] = hid + (size_t)srow * HH + seg * 8;
        alds[it]  = row * KPAD + seg * 8;
    }
    auto async_a = [&](int kk, int buf) {
        #pragma unroll
        for (int it = 0; it < 2; ++it) {
            unsigned lds = (unsigned)(uintptr_t)(&As[buf][alds[it]]);
            unsigned long long ga = (unsigned long long)(uintptr_t)(agsrc[it] + kk);
            asm volatile("global_load_async_to_lds_b128 %0, %1, off"
                         :: "v"(lds), "v"(ga) : "memory");
        }
    };

    // B staging: 1024 slots = 32 k x 32 float4, transposed + converted.
    const float* brow[4];
    int bk[4], bn[4];
    #pragma unroll
    for (int it = 0; it < 4; ++it) {
        int slot = threadIdx.x + it * 256;
        bk[it] = slot >> 5;
        bn[it] = (slot & 31) * 4;
        brow[it] = W + (size_t)bk[it] * AA + n0 + bn[it];
    }
    float4 fb[4];
    auto load_b = [&](int kk) {
        #pragma unroll
        for (int it = 0; it < 4; ++it) fb[it] = *(const float4*)(brow[it] + (size_t)kk * AA);
    };
    auto store_b = [&](int buf) {
        #pragma unroll
        for (int it = 0; it < 4; ++it) {
            Bs[buf][(bn[it] + 0) * KPAD + bk[it]] = f2bf(fb[it].x);
            Bs[buf][(bn[it] + 1) * KPAD + bk[it]] = f2bf(fb[it].y);
            Bs[buf][(bn[it] + 2) * KPAD + bk[it]] = f2bf(fb[it].z);
            Bs[buf][(bn[it] + 3) * KPAD + bk[it]] = f2bf(fb[it].w);
        }
    };

    v8f acc[8] = {};

    // prologue: stage tile kk=0 into buffer 0
    async_a(0, 0);
    load_b(0);
    store_b(0);
    asm volatile("s_wait_asynccnt 0" ::: "memory");   // A DMA landed in LDS

    int cur = 0;
    for (int kk = 0; kk < HH; kk += TILE_K) {
        const bool more = (kk + TILE_K) < HH;
        __syncthreads();                              // publish buf[cur]
        if (more) {
            async_a(kk + TILE_K, cur ^ 1);            // DMA next A tile during compute
            load_b(kk + TILE_K);
        }

        Frag16 a;
        a.u4[0] = *(const uint4*)(&As[cur][(wave * 16 + r) * KPAD + ksel]);
        a.u4[1] = *(const uint4*)(&As[cur][(wave * 16 + r) * KPAD + 16 + ksel]);
        #pragma unroll
        for (int ns = 0; ns < 8; ++ns) {
            Frag16 b;
            b.u4[0] = *(const uint4*)(&Bs[cur][(ns * 16 + r) * KPAD + ksel]);
            b.u4[1] = *(const uint4*)(&Bs[cur][(ns * 16 + r) * KPAD + 16 + ksel]);
            acc[ns] = wmma_bf16(a, b, acc[ns]);
        }
        if (more) store_b(cur ^ 1);
        asm volatile("s_wait_asynccnt 0" ::: "memory");  // A DMA done before next publish
        cur ^= 1;
    }

    // epilogue: bias, scatter rows to out via idx
    #pragma unroll
    for (int ns = 0; ns < 8; ++ns) {
        int ncol = n0 + ns * 16 + r;
        float bias = b2[(size_t)e * AA + ncol];
        #pragma unroll
        for (int j = 0; j < 8; ++j) {
            int ml = wave * 16 + half * 8 + j;
            if (ml < valid) {
                int token = idx[m0 + ml];
                out[(size_t)token * AA + ncol] = acc[ns][j] + bias;
            }
        }
    }
}

// ---------------------------------------------------------------------------
extern "C" void kernel_launch(void* const* d_in, const int* in_sizes, int n_in,
                              void* d_out, int out_size, void* d_ws, size_t ws_size,
                              hipStream_t stream) {
    const float*     h    = (const float*)d_in[0];
    const long long* env  = (const long long*)d_in[1];   // jnp.int64
    const float*     W1   = (const float*)d_in[2];
    const float*     b1   = (const float*)d_in[3];
    const float*     W2   = (const float*)d_in[4];
    const float*     b2   = (const float*)d_in[5];
    float*           out  = (float*)d_out;

    unsigned char*  ws    = (unsigned char*)d_ws;
    unsigned*       ws_u  = (unsigned*)ws;
    const unsigned* offs  = (const unsigned*)(ws + WS_OFFS);
    int*            idx   = (int*)(ws + WS_IDX);
    __bf16*         hid   = (__bf16*)(ws + WS_HID);

    (void)in_sizes; (void)n_in; (void)out_size; (void)ws_size;

    // 1) bucket tokens by env (counts -> prefix -> scatter); re-done every call
    k_zero<<<1, 32, 0, stream>>>(ws_u);
    k_count<<<BB / 256, 256, 0, stream>>>(env, ws_u);
    k_scan<<<1, 1, 0, stream>>>(ws_u);
    k_scatter<<<BB / 256, 256, 0, stream>>>(env, ws_u, idx);

    // 2) hid = relu(h_sorted @ W1[e] + b1[e])  (bf16 WMMA)
    k_gemm1<<<dim3(MAX_MTILES, HH / TILE_N), 256, 0, stream>>>(h, W1, b1, offs, idx, hid);

    // 3) out = scatter(hid @ W2[e] + b2[e])    (bf16 WMMA + async-LDS A tiles)
    k_gemm2<<<dim3(MAX_MTILES, AA / TILE_N), 256, 0, stream>>>(hid, W2, b2, offs, idx, out);
}
